// MoEBlock_19859928776970
// MI455X (gfx1250) — matrix operand order repaired and checked by reading, not compiled, not run on hardware
//
#include <hip/hip_runtime.h>

typedef __attribute__((ext_vector_type(16))) __bf16 v16bf;
typedef __attribute__((ext_vector_type(8)))  __bf16 v8bf;
typedef __attribute__((ext_vector_type(8)))  float  v8f;

#define B_   32
#define C_   256
#define H_   64
#define W_   64
#define E_   8
#define KDIM 2304     // C_ * 9, K-order: (kh*3+kw)*256 + ci
#define PW   66       // padded width/height (image + 1 halo each side)
#define PATCH (3 * PW * 32)   // one staged chunk: 3 rows x 66 cols x 32 ci (bf16)

// ---------------- kernel 1: global average pool -> pooled[B][C] ----------------
__global__ __launch_bounds__(256)
void pool_kernel(const float* __restrict__ x, float* __restrict__ pooled) {
    const int c = blockIdx.x, b = blockIdx.y, t = threadIdx.x;
    const float* px = x + (((size_t)b * C_ + c) << 12);   // 4096 contiguous
    float s = 0.f;
    #pragma unroll
    for (int i = 0; i < 16; ++i) s += px[t + (i << 8)];
    __shared__ float red[256];
    red[t] = s; __syncthreads();
    for (int o = 128; o > 0; o >>= 1) {
        if (t < o) red[t] += red[t + o];
        __syncthreads();
    }
    if (t == 0) pooled[b * C_ + c] = red[0] * (1.0f / 4096.0f);
}

// ---------------- kernel 2: gate -> softmax -> top2 -> renorm -> comb[B][E] ----------------
__global__ void gate_kernel(const float* __restrict__ pooled,
                            const float* __restrict__ gw,
                            const float* __restrict__ gb,
                            float* __restrict__ comb) {
    const int b = threadIdx.x;
    if (b >= B_) return;
    float logit[E_];
    for (int e = 0; e < E_; ++e) {
        float s = gb[e];
        const float* p = pooled + b * C_;
        const float* w = gw + e * C_;
        for (int c = 0; c < C_; ++c) s += p[c] * w[c];
        logit[e] = s;
    }
    float m = logit[0];
    for (int e = 1; e < E_; ++e) m = fmaxf(m, logit[e]);
    float wgt[E_]; float den = 0.f;
    for (int e = 0; e < E_; ++e) { wgt[e] = __expf(logit[e] - m); den += wgt[e]; }
    float inv = 1.f / den;
    for (int e = 0; e < E_; ++e) wgt[e] *= inv;
    int i1 = 0; float v1 = wgt[0];
    for (int e = 1; e < E_; ++e) if (wgt[e] > v1) { v1 = wgt[e]; i1 = e; }
    int i2 = -1; float v2 = -1.f;
    for (int e = 0; e < E_; ++e) if (e != i1 && wgt[e] > v2) { v2 = wgt[e]; i2 = e; }
    float t2 = __expf(v2 - v1);
    float is = 1.f / (1.f + t2);
    float o[E_];
    for (int e = 0; e < E_; ++e) o[e] = 0.f;
    o[i1] = is; o[i2] = t2 * is;
    for (int e = 0; e < E_; ++e) comb[b * E_ + e] = o[e];
}

// ---------------- kernel 3: merge expert weights/bias per batch, cast to bf16 ----------------
__global__ __launch_bounds__(256)
void merge_kernel(const float* __restrict__ conv_w,
                  const float* __restrict__ conv_b,
                  const float* __restrict__ comb,
                  unsigned short* __restrict__ Wm_u,
                  float* __restrict__ biasW) {
    __bf16* Wm = reinterpret_cast<__bf16*>(Wm_u);
    const int co = blockIdx.x, b = blockIdx.y, t = threadIdx.x;
    float cb[E_];
    #pragma unroll
    for (int e = 0; e < E_; ++e) cb[e] = comb[b * E_ + e];
    for (int k = t; k < KDIM; k += 256) {
        const int ci = k & 255, pos = k >> 8;
        float s = 0.f;
        #pragma unroll
        for (int e = 0; e < E_; ++e)
            s += cb[e] * conv_w[(((size_t)e * C_ + co) * C_ + ci) * 9 + pos];
        Wm[((size_t)(b * C_ + co)) * KDIM + k] = (__bf16)s;
    }
    if (t == 0) {
        float s = 0.f;
        #pragma unroll
        for (int e = 0; e < E_; ++e) s += cb[e] * conv_b[e * C_ + co];
        biasW[b * C_ + co] = s;
    }
}

// ---------------- kernel 3b: pre-pad + convert x -> bf16 xpad[b][ci][66][66] ----------------
__global__ __launch_bounds__(256)
void pad_kernel(const float* __restrict__ x, unsigned short* __restrict__ xpad_u) {
    __bf16* xpad = reinterpret_cast<__bf16*>(xpad_u);
    const int ci = blockIdx.x, b = blockIdx.y, t = threadIdx.x;
    const float* src = x + (((size_t)b * C_ + ci) << 12);
    __bf16* dst = xpad + ((size_t)(b * C_ + ci)) * (PW * PW);
    for (int j = t; j < PW * PW; j += 256) {
        const int pc = j % PW, pr = j / PW;
        const int ir = pr - 1, ic = pc - 1;
        float v = 0.f;
        if (ir >= 0 && ir < H_ && ic >= 0 && ic < W_)
            v = src[(ir << 6) + ic];
        dst[j] = (__bf16)v;
    }
}

// ---------------- kernel 4: implicit-GEMM 3x3 conv via bf16 WMMA ----------------
// grid: (H rows, C_out/128, B). block: 256 threads = 8 waves.
// wave w: 16 C_out rows x 64 pixels (4 N-subtiles of 16), K = 2304 in 72 chunks of 32.
// LDS double buffer: stage chunk cc+1 while computing chunk cc (one barrier/chunk).
__global__ __launch_bounds__(256)
void conv_kernel(const unsigned short* __restrict__ xpad_u,
                 const unsigned short* __restrict__ Wm_u,
                 const float* __restrict__ biasW,
                 float* __restrict__ out) {
    const __bf16* xpad = reinterpret_cast<const __bf16*>(xpad_u);
    const __bf16* Wm   = reinterpret_cast<const __bf16*>(Wm_u);
    const int r    = blockIdx.x;            // output row
    const int co0  = blockIdx.y * 128;
    const int b    = blockIdx.z;
    const int t    = threadIdx.x;
    const int lane = t & 31;
    const int wv   = t >> 5;
    const int ln16 = lane & 15;
    const bool hiH = lane >= 16;
    const int coBase = co0 + wv * 16;

    // double-buffered patch: [buf][row][col][ci], ci contiguous (B-fragment order)
    __shared__ __attribute__((aligned(16))) __bf16 Xs[2][PATCH];

    v8f acc[4] = {v8f{}, v8f{}, v8f{}, v8f{}};

    // padded rows r..r+2 correspond to image rows r-1..r+1
    const __bf16* xb = xpad + (((size_t)b * C_) * PW + r) * PW;
    const __bf16* Arow0 = Wm + ((size_t)(b * C_ + coBase + ln16)) * KDIM;

    auto stage = [&](int cc, int buf) {
        const __bf16* src = xb + ((size_t)cc * 32) * (PW * PW);
        for (int i = t; i < PATCH; i += 256) {
            const int col   = i % PW;          // global-contiguous along col
            const int rowci = i / PW;
            const int row   = rowci % 3;
            const int ci    = rowci / 3;
            Xs[buf][((row * PW) + col) * 32 + ci] =
                src[(size_t)ci * (PW * PW) + row * PW + col];
        }
    };

    stage(0, 0);
    __syncthreads();

    for (int cc = 0; cc < 8; ++cc) {
        const int buf = cc & 1;
        if (cc < 7) stage(cc + 1, buf ^ 1);   // overlaps with compute below

        const __bf16* Arow = Arow0 + cc * 32;
        if (cc < 7) __builtin_prefetch(Arow0 + (cc + 1) * 32, 0, 0);  // global_prefetch_b8

        #pragma unroll
        for (int pos = 0; pos < 9; ++pos) {
            const int kh = pos / 3;
            const int kw = pos - kh * 3;
            // A fragment: lane<16 holds K {0..7,16..23}, lane>=16 holds K {8..15,24..31}
            const __bf16* ap = Arow + pos * 256 + (hiH ? 8 : 0);
            v8bf alo = *(const v8bf*)(ap);
            v8bf ahi = *(const v8bf*)(ap + 16);
            v16bf a = __builtin_shufflevector(alo, ahi,
                        0, 1, 2, 3, 4, 5, 6, 7, 8, 9, 10, 11, 12, 13, 14, 15);

            const int ciOff = hiH ? 8 : 0;  // B: lanes 0-15 -> K {0..7,16..23}, lanes 16-31 -> K {8..15,24..31}
            #pragma unroll
            for (int s = 0; s < 4; ++s) {
                const int p = s * 16 + ln16;                 // pixel column
                const __bf16* bp = &Xs[buf][((kh * PW) + p + kw) * 32 + ciOff];
                v8bf blo = *(const v8bf*)bp;
                v8bf bhi = *(const v8bf*)(bp + 16);
                v16bf bb = __builtin_shufflevector(blo, bhi,
                            0, 1, 2, 3, 4, 5, 6, 7, 8, 9, 10, 11, 12, 13, 14, 15);
                acc[s] = __builtin_amdgcn_wmma_f32_16x16x32_bf16(
                            false, a, false, bb, (short)0, acc[s], false, false);
            }
        }
        __syncthreads();   // staging of next buffer complete + this buffer free
    }

    // epilogue: D layout — VGPR j: lanes 0-15 M=j, lanes 16-31 M=j+8; N = lane%16
    const int mOff = hiH ? 8 : 0;
    float bj[8];
    #pragma unroll
    for (int j = 0; j < 8; ++j) bj[j] = biasW[b * C_ + coBase + mOff + j];
    #pragma unroll
    for (int s = 0; s < 4; ++s) {
        const int pix = s * 16 + ln16;
        #pragma unroll
        for (int j = 0; j < 8; ++j) {
            const int co = coBase + mOff + j;
            out[((size_t)(b * C_ + co)) * 4096 + r * 64 + pix] = acc[s][j] + bj[j];
        }
    }
}

extern "C" void kernel_launch(void* const* d_in, const int* in_sizes, int n_in,
                              void* d_out, int out_size, void* d_ws, size_t ws_size,
                              hipStream_t stream) {
    (void)in_sizes; (void)n_in; (void)out_size; (void)ws_size;
    const float* x      = (const float*)d_in[0];
    const float* gate_w = (const float*)d_in[1];
    const float* gate_b = (const float*)d_in[2];
    const float* conv_w = (const float*)d_in[3];
    const float* conv_b = (const float*)d_in[4];
    float* out = (float*)d_out;

    char* ws = (char*)d_ws;
    float*          pooled = (float*)(ws);                  // 32*256*4        = 32768 B
    float*          comb   = (float*)(ws + 32768);          // 32*8*4          = 1024 B
    float*          biasW  = (float*)(ws + 33792);          // 32*256*4        = 32768 B
    unsigned short* Wm     = (unsigned short*)(ws + 66560); // 32*256*2304*2   = 37748736 B
    unsigned short* xpad   = (unsigned short*)(ws + 37815296); // 32*256*66*66*2 = 71368704 B

    pool_kernel <<<dim3(C_, B_), 256, 0, stream>>>(x, pooled);
    gate_kernel <<<1, 32, 0, stream>>>(pooled, gate_w, gate_b, comb);
    merge_kernel<<<dim3(C_, B_), 256, 0, stream>>>(conv_w, conv_b, comb, Wm, biasW);
    pad_kernel  <<<dim3(C_, B_), 256, 0, stream>>>(x, xpad);
    conv_kernel <<<dim3(H_, C_ / 128, B_), 256, 0, stream>>>(xpad, Wm, biasW, out);
}
